// CapsuleNetwork_27522150433329
// MI455X (gfx1250) — matrix operand order repaired and checked by reading, not compiled, not run on hardware
//
#include <hip/hip_runtime.h>
#include <hip/hip_bf16.h>
#include <cstdint>
#include <math.h>

// ---------------------------------------------------------------------------
// CDNA5 (gfx1250) CapsNet forward. GEMM-shaped work on v_wmma_f32_16x16x32_bf16
// (wave32). 64x64 block tiles, 8 waves x two 16x16 C tiles, double-buffered
// LDS, all-b128 LDS traffic: weights are pre-transposed (NxK) + zero-padded in
// one-time prep kernels, so the hot loop has no transpose stores and no guards.
// ---------------------------------------------------------------------------

typedef __attribute__((ext_vector_type(16))) __bf16 v16bf;
typedef __attribute__((ext_vector_type(8)))  __bf16 v8bf;
typedef __attribute__((ext_vector_type(8)))  float  v8f;
typedef unsigned short bf16raw;   // host-safe storage type for bf16 buffers

#define LDT 40      // padded LDS row length (bf16): 80B rows -> stride-4 banks
#define EPSF 1e-7f

union Frag { v16bf v; v8bf h[2]; };

// A fragment (16x32 bf16, ISA 7.12.2): lane m=lane&15; per lane the 16 elems
// are two contiguous 8-runs: k = hi*8 + [0..7] and k = 16 + hi*8 + [0..7].
__device__ __forceinline__ v16bf load_a_frag(const __bf16* As, int m, int lane) {
  Frag f;
  const __bf16* p = As + m * LDT + (((lane >> 4) & 1) << 3);
  f.h[0] = *(const v8bf*)p;
  f.h[1] = *(const v8bf*)(p + 16);
  return f.v;
}
// B fragment (32x16 bf16): lane n=lane&15; k = hi*8 + [0..7] and k = 16 + hi*8 +
// [0..7] in the n-major LDS tile Bt[n][k] (same order as A).
__device__ __forceinline__ v16bf load_b_frag(const __bf16* Bt, int n, int lane) {
  Frag f;
  const __bf16* p = Bt + n * LDT + (((lane >> 4) & 1) << 3);
  f.h[0] = *(const v8bf*)p;
  f.h[1] = *(const v8bf*)(p + 16);
  return f.v;
}

#define GEMM_PROLOG                                                     \
  const int tid = threadIdx.x, lane = tid & 31, wave = tid >> 5;        \
  const int tc = wave & 3, trP = wave >> 2;                             \
  const int m0 = blockIdx.x * 64, n0 = blockIdx.y * 64;                 \
  const int ar = tid >> 2, ac = (tid & 3) << 3; /* A: row, 8 k/thread */\
  const int bn = tid >> 2, bk = (tid & 3) << 3; /* B^T: n-row, 8 k/thr */\
  v8f acc0 = {}, acc1 = {};

#define GEMM_COMPUTE(buf)                                               \
  {                                                                     \
    v16bf bf = load_b_frag(&Bt[buf][0], tc * 16 + (lane & 15), lane);   \
    v16bf a0 = load_a_frag(&As[buf][0], trP * 32 + (lane & 15), lane);  \
    v16bf a1 = load_a_frag(&As[buf][0], trP * 32 + 16 + (lane & 15), lane); \
    acc0 = __builtin_amdgcn_wmma_f32_16x16x32_bf16(false, a0, false, bf, \
                                                   (short)0, acc0, false, false); \
    acc1 = __builtin_amdgcn_wmma_f32_16x16x32_bf16(false, a1, false, bf, \
                                                   (short)0, acc1, false, false); \
  }

// --- generic GEMM: C = act(scale*(A@BT^T) + bias) ---------------------------
// A[MxK] bf16 row-major, BT[Npad x K] bf16 (pre-transposed weights, rows >= N
// zero-filled). M%64==0, K%32==0, grid.y*64 <= Npad.
__global__ __launch_bounds__(256) void gemm_bf16_wmma(
    const bf16raw* __restrict__ Araw, const bf16raw* __restrict__ BTraw,
    const float* __restrict__ bias, float* __restrict__ outF,
    bf16raw* __restrict__ outHraw, int M, int N, int K, float scale, int act) {
  const __bf16* A  = (const __bf16*)Araw;
  const __bf16* BT = (const __bf16*)BTraw;
  __bf16* outH = (__bf16*)outHraw;
  __shared__ __align__(16) __bf16 As[2][64 * LDT];
  __shared__ __align__(16) __bf16 Bt[2][64 * LDT];
  GEMM_PROLOG

  auto stage = [&](int k0, int buf) {
    v8bf va = *(const v8bf*)(A + (size_t)(m0 + ar) * K + k0 + ac);
    *(v8bf*)&As[buf][ar * LDT + ac] = va;
    v8bf vb = *(const v8bf*)(BT + (size_t)(n0 + bn) * K + k0 + bk);
    *(v8bf*)&Bt[buf][bn * LDT + bk] = vb;
  };

  stage(0, 0);
  __syncthreads();
  int buf = 0;
  for (int k0 = 0; k0 < K; k0 += 32) {
    if (k0 + 32 < K) {
      stage(k0 + 32, buf ^ 1);
      if (k0 + 64 < K)
        __builtin_prefetch((const void*)(BT + (size_t)(n0 + bn) * K + k0 + 64), 0, 3);
    }
    GEMM_COMPUTE(buf)
    __syncthreads();
    buf ^= 1;
  }

  const int hi = (lane >> 4) & 1;
  const int nn = n0 + tc * 16 + (lane & 15);
#pragma unroll
  for (int r = 0; r < 8; ++r) {
    int mA = m0 + trP * 32 + hi * 8 + r;
#pragma unroll
    for (int half = 0; half < 2; ++half) {
      int m = mA + half * 16;
      float v = (half ? acc1[r] : acc0[r]) * scale + (bias ? bias[nn] : 0.0f);
      if (act == 1)      v = fmaxf(v, 0.0f);
      else if (act == 2) v = 1.0f / (1.0f + __expf(-v));
      if (m < M && nn < N) {
        if (outF) outF[(size_t)m * N + nn] = v;
        if (outH) outH[(size_t)m * N + nn] = (__bf16)v;
      }
    }
  }
}

// --- conv1: implicit GEMM, M=B*400=102400, N=256, K=81 (padded 96) ----------
// Weights pre-transposed+padded: c1wT[256][96].
__global__ __launch_bounds__(256) void conv1_wmma(
    const float* __restrict__ X, const bf16raw* __restrict__ WTraw,
    const float* __restrict__ bias, bf16raw* __restrict__ outRaw) {
  const __bf16* WT = (const __bf16*)WTraw;
  __bf16* out = (__bf16*)outRaw;
  __shared__ __align__(16) __bf16 As[2][64 * LDT];
  __shared__ __align__(16) __bf16 Bt[2][64 * LDT];
  GEMM_PROLOG
  const int gr = m0 + ar;
  const int b = gr / 400, p = gr - b * 400, oy = p / 20, ox = p - oy * 20;
  const float* Xrow = X + (size_t)b * 784 + (size_t)oy * 28 + ox;

  auto stage = [&](int k0, int buf) {
    v8bf va;
#pragma unroll
    for (int i = 0; i < 8; ++i) {
      int k = k0 + ac + i;
      int kk = k < 81 ? k : 80;              // clamp -> always-valid address
      int ky = kk / 9, kx = kk - ky * 9;
      float x = Xrow[ky * 28 + kx];
      va[i] = (__bf16)(k < 81 ? x : 0.0f);   // v_cndmask, no exec juggling
    }
    *(v8bf*)&As[buf][ar * LDT + ac] = va;
    v8bf vb = *(const v8bf*)(WT + (size_t)(n0 + bn) * 96 + k0 + bk);
    *(v8bf*)&Bt[buf][bn * LDT + bk] = vb;
  };

  stage(0, 0);
  __syncthreads();
  int buf = 0;
  for (int k0 = 0; k0 < 96; k0 += 32) {
    if (k0 + 32 < 96) stage(k0 + 32, buf ^ 1);
    GEMM_COMPUTE(buf)
    __syncthreads();
    buf ^= 1;
  }

  const int hi = (lane >> 4) & 1;
  const int nn = n0 + tc * 16 + (lane & 15);
  const float bs = bias[nn];
#pragma unroll
  for (int r = 0; r < 8; ++r) {
    int mA = m0 + trP * 32 + hi * 8 + r;
    out[(size_t)mA * 256 + nn]        = (__bf16)fmaxf(acc0[r] + bs, 0.0f);
    out[(size_t)(mA + 16) * 256 + nn] = (__bf16)fmaxf(acc1[r] + bs, 0.0f);
  }
}

// --- conv2: implicit GEMM, M=B*36=9216, N=256, K=20736, stride 2 ------------
// Weights pre-transposed: c2wT[256][20736].
__global__ __launch_bounds__(256) void conv2_wmma(
    const bf16raw* __restrict__ Ainraw, const bf16raw* __restrict__ WTraw,
    const float* __restrict__ bias, float* __restrict__ out) {
  const __bf16* Ain = (const __bf16*)Ainraw;   // conv1 out (B,20,20,256) bf16
  const __bf16* WT  = (const __bf16*)WTraw;
  __shared__ __align__(16) __bf16 As[2][64 * LDT];
  __shared__ __align__(16) __bf16 Bt[2][64 * LDT];
  GEMM_PROLOG
  const int gr = m0 + ar;
  const int b = gr / 36, p = gr - b * 36, oy = p / 6, ox = p - oy * 6;
  const __bf16* Abase = Ain + ((size_t)(b * 20 + oy * 2) * 20 + ox * 2) * 256;

  auto stage = [&](int k0, int buf) {
    // an 8-run of k stays inside one (ky,kx) tap (k0%32==0, 256%32==0)
    int k = k0 + ac;
    int t = k >> 8, cin = k & 255;
    int ky = t / 9, kx = t - ky * 9;
    v8bf va = *(const v8bf*)(Abase + ((size_t)(ky * 20 + kx)) * 256 + cin);
    *(v8bf*)&As[buf][ar * LDT + ac] = va;
    v8bf vb = *(const v8bf*)(WT + (size_t)(n0 + bn) * 20736 + k0 + bk);
    *(v8bf*)&Bt[buf][bn * LDT + bk] = vb;
  };

  stage(0, 0);
  __syncthreads();
  int buf = 0;
  for (int k0 = 0; k0 < 20736; k0 += 32) {
    if (k0 + 32 < 20736) {
      stage(k0 + 32, buf ^ 1);
      if (k0 + 64 < 20736)
        __builtin_prefetch((const void*)(WT + (size_t)(n0 + bn) * 20736 + k0 + 64), 0, 3);
    }
    GEMM_COMPUTE(buf)
    __syncthreads();
    buf ^= 1;
  }

  const int hi = (lane >> 4) & 1;
  const int nn = n0 + tc * 16 + (lane & 15);
  const float bs = bias[nn];
#pragma unroll
  for (int r = 0; r < 8; ++r) {
    int mA = m0 + trP * 32 + hi * 8 + r;
    out[(size_t)mA * 256 + nn]        = fmaxf(acc0[r] + bs, 0.0f);
    out[(size_t)(mA + 16) * 256 + nn] = fmaxf(acc1[r] + bs, 0.0f);
  }
}

// --- prep kernels -----------------------------------------------------------
// dst[Npad x Kpad] = transpose(src[Ksrc x Nsrc]) as bf16, zero-padded.
__global__ void transpose_pad_bf16(const float* __restrict__ src, bf16raw* __restrict__ dst,
                                   int Ksrc, int Nsrc, int Kpad, int Npad) {
  int o = blockIdx.x * 256 + threadIdx.x;
  if (o < Npad * Kpad) {
    int n = o / Kpad, k = o - n * Kpad;
    float v = (n < Nsrc && k < Ksrc) ? src[(size_t)k * Nsrc + n] : 0.0f;
    ((__bf16*)dst)[o] = (__bf16)v;
  }
}

// W (1152,10,16,8) -> WtT[(c*16+i)][(n*8+d)] bf16, padded to 192 rows x 9216
__global__ void transpose_W(const float* __restrict__ W, bf16raw* __restrict__ WtT) {
  int o = blockIdx.x * 256 + threadIdx.x;
  if (o < 192 * 9216) {
    int row = o / 9216, k = o - row * 9216;
    float v = 0.0f;
    if (row < 160) {
      int n = k >> 3, d = k & 7, c = row >> 4, ii = row & 15;
      v = W[(((size_t)(n * 10 + c) * 16) + ii) * 8 + d];
    }
    ((__bf16*)WtT)[o] = (__bf16)v;
  }
}

// caps1 = squash over groups of 8; write f32 + bf16 copies
__global__ void squash_caps1(const float* __restrict__ h, float* __restrict__ cf,
                             bf16raw* __restrict__ ch) {
  int g = blockIdx.x * 256 + threadIdx.x;
  if (g < 256 * 1152) {
    const float* s = h + (size_t)g * 8;
    float v[8], sq = 0.0f;
#pragma unroll
    for (int j = 0; j < 8; ++j) { v[j] = s[j]; sq += v[j] * v[j]; }
    float f = (sq / (1.0f + sq)) / sqrtf(sq + EPSF);
#pragma unroll
    for (int j = 0; j < 8; ++j) {
      cf[(size_t)g * 8 + j] = v[j] * f;
      ((__bf16*)ch)[(size_t)g * 8 + j] = (__bf16)(v[j] * f);
    }
  }
}

// v1 = squash(s1) over last dim (16)
__global__ void squash_v1(const float* __restrict__ s1, float* __restrict__ v1) {
  int g = blockIdx.x * 256 + threadIdx.x;
  if (g < 256 * 10) {
    const float* s = s1 + (size_t)g * 16;
    float sq = 0.0f;
    for (int j = 0; j < 16; ++j) sq += s[j] * s[j];
    float f = (sq / (1.0f + sq)) / sqrtf(sq + EPSF);
    for (int j = 0; j < 16; ++j) v1[(size_t)g * 16 + j] = s[j] * f;
  }
}

// --- fused routing: recompute u_hat per (b,n); b2 -> softmax -> s2 -> v2 ----
__global__ __launch_bounds__(160) void routing_kernel(
    const float* __restrict__ caps1, const float* __restrict__ W,
    const float* __restrict__ v1, const int* __restrict__ Y,
    float* __restrict__ outCaps2, float* __restrict__ outPred,
    bf16raw* __restrict__ decIn) {
  __shared__ float c1s[9216];
  __shared__ float v1s[160], s2s[160], prod[160];
  __shared__ float b2s[10], r2s[10], sqs[10];
  const int b = blockIdx.x, t = threadIdx.x;
  for (int i = t; i < 9216; i += 160) c1s[i] = caps1[(size_t)b * 9216 + i];
  v1s[t] = v1[(size_t)b * 160 + t];
  s2s[t] = 0.0f;
  __syncthreads();
  const int c = t >> 4, ii = t & 15;
  for (int n = 0; n < 1152; ++n) {
    const float* wr = W + (((size_t)(n * 10 + c) * 16) + ii) * 8;
    const float* cr = c1s + n * 8;
    float u = 0.0f;
#pragma unroll
    for (int d = 0; d < 8; ++d) u += wr[d] * cr[d];
    prod[t] = u * v1s[t];
    __syncthreads();
    if (t < 10) {
      float s = 0.0f;
      for (int j = 0; j < 16; ++j) s += prod[t * 16 + j];
      b2s[t] = s;
    }
    __syncthreads();
    if (t == 0) {
      float mx = b2s[0];
      for (int q = 1; q < 10; ++q) mx = fmaxf(mx, b2s[q]);
      float ss = 0.0f;
      for (int q = 0; q < 10; ++q) { float e = __expf(b2s[q] - mx); r2s[q] = e; ss += e; }
      float inv = 1.0f / ss;
      for (int q = 0; q < 10; ++q) r2s[q] *= inv;
    }
    __syncthreads();
    s2s[t] += r2s[c] * u;
    __syncthreads();
  }
  prod[t] = s2s[t] * s2s[t];
  __syncthreads();
  if (t < 10) {
    float s = 0.0f;
    for (int j = 0; j < 16; ++j) s += prod[t * 16 + j];
    sqs[t] = s;
  }
  __syncthreads();
  float sq = sqs[c];
  float v2 = (sq / (1.0f + sq)) * (s2s[t] / sqrtf(sq + EPSF));
  outCaps2[(size_t)b * 160 + t] = v2;
  int y = Y[b];
  ((__bf16*)decIn)[(size_t)b * 160 + t] = (__bf16)((c == y) ? v2 : 0.0f);
  if (t == 0) {
    int best = 0; float bv = sqs[0];
    for (int q = 1; q < 10; ++q) if (sqs[q] > bv) { bv = sqs[q]; best = q; }
    outPred[b] = (float)best;
  }
}

// ---------------------------------------------------------------------------
extern "C" void kernel_launch(void* const* d_in, const int* in_sizes, int n_in,
                              void* d_out, int out_size, void* d_ws, size_t ws_size,
                              hipStream_t stream) {
  (void)in_sizes; (void)n_in; (void)out_size; (void)ws_size;
  const float* X   = (const float*)d_in[0];
  const int*   Y   = (const int*)d_in[1];
  const float* c1w = (const float*)d_in[2];
  const float* c1b = (const float*)d_in[3];
  const float* c2w = (const float*)d_in[4];
  const float* c2b = (const float*)d_in[5];
  const float* W   = (const float*)d_in[6];
  const float* d1w = (const float*)d_in[7];
  const float* d1b = (const float*)d_in[8];
  const float* d2w = (const float*)d_in[9];
  const float* d2b = (const float*)d_in[10];
  const float* d3w = (const float*)d_in[11];
  const float* d3b = (const float*)d_in[12];

  float* out       = (float*)d_out;
  float* outCaps2  = out;                     // 256*160
  float* outDec    = out + 40960;             // 256*784
  float* outPred   = out + 40960 + 200704;    // 256

  uint8_t* wsp = (uint8_t*)d_ws;
  auto take = [&](size_t bytes) -> void* {
    void* p = (void*)wsp;
    wsp += (bytes + 255) & ~(size_t)255;
    return p;
  };
  // pre-transposed (+padded) bf16 weights
  bf16raw* c1wT    = (bf16raw*)take((size_t)256 * 96 * 2);
  bf16raw* c2wT    = (bf16raw*)take((size_t)256 * 20736 * 2);
  bf16raw* d1wT    = (bf16raw*)take((size_t)512 * 160 * 2);
  bf16raw* d2wT    = (bf16raw*)take((size_t)1024 * 512 * 2);
  bf16raw* d3wT    = (bf16raw*)take((size_t)832 * 1024 * 2);
  bf16raw* WtT     = (bf16raw*)take((size_t)192 * 9216 * 2);
  // activations / intermediates
  bf16raw* conv1_h = (bf16raw*)take((size_t)102400 * 256 * 2);
  float*   conv2_f = (float*)take((size_t)9216 * 256 * 4);
  float*   caps1_f = (float*)take((size_t)9216 * 256 * 4);
  bf16raw* caps1_h = (bf16raw*)take((size_t)9216 * 256 * 2);
  float*   s1_f    = (float*)take((size_t)256 * 160 * 4);
  float*   v1_f    = (float*)take((size_t)256 * 160 * 4);
  bf16raw* decin_h = (bf16raw*)take((size_t)256 * 160 * 2);
  bf16raw* h1_h    = (bf16raw*)take((size_t)256 * 512 * 2);
  bf16raw* h2_h    = (bf16raw*)take((size_t)256 * 1024 * 2);

  auto cdiv = [](int a, int b) { return (a + b - 1) / b; };

  // 1) weight conversion: transpose to NxK bf16 with zero padding
  transpose_pad_bf16<<<cdiv(256 * 96, 256), 256, 0, stream>>>(c1w, c1wT, 81, 256, 96, 256);
  transpose_pad_bf16<<<cdiv(256 * 20736, 256), 256, 0, stream>>>(c2w, c2wT, 20736, 256, 20736, 256);
  transpose_pad_bf16<<<cdiv(512 * 160, 256), 256, 0, stream>>>(d1w, d1wT, 160, 512, 160, 512);
  transpose_pad_bf16<<<cdiv(1024 * 512, 256), 256, 0, stream>>>(d2w, d2wT, 512, 1024, 512, 1024);
  transpose_pad_bf16<<<cdiv(832 * 1024, 256), 256, 0, stream>>>(d3w, d3wT, 1024, 784, 1024, 832);
  transpose_W<<<cdiv(192 * 9216, 256), 256, 0, stream>>>(W, WtT);

  // 2) conv1 (WMMA implicit GEMM): M=102400, N=256
  conv1_wmma<<<dim3(1600, 4), 256, 0, stream>>>(X, c1wT, c1b, conv1_h);

  // 3) conv2 (WMMA implicit GEMM): M=9216, N=256, K=20736 — dominant cost
  conv2_wmma<<<dim3(144, 4), 256, 0, stream>>>(conv1_h, c2wT, c2b, conv2_f);

  // 4) caps1 = squash(groups of 8)
  squash_caps1<<<cdiv(294912, 256), 256, 0, stream>>>(conv2_f, caps1_f, caps1_h);

  // 5) s1 = (caps1 @ Wt) / 10  (WMMA GEMM, M=256, N=160 (pad 192), K=9216)
  gemm_bf16_wmma<<<dim3(4, 3), 256, 0, stream>>>(
      caps1_h, WtT, nullptr, s1_f, nullptr, 256, 160, 9216, 0.1f, 0);

  // 6) v1 = squash(s1)
  squash_v1<<<cdiv(2560, 256), 256, 0, stream>>>(s1_f, v1_f);

  // 7) fused routing (u_hat recomputed; never hits HBM) + v2/argmax/mask
  routing_kernel<<<256, 160, 0, stream>>>(caps1_f, W, v1_f, Y,
                                          outCaps2, outPred, decin_h);

  // 8) decoder: 3 WMMA GEMMs (relu, relu, sigmoid)
  gemm_bf16_wmma<<<dim3(4, 8), 256, 0, stream>>>(
      decin_h, d1wT, d1b, nullptr, h1_h, 256, 512, 160, 1.0f, 1);
  gemm_bf16_wmma<<<dim3(4, 16), 256, 0, stream>>>(
      h1_h, d2wT, d2b, nullptr, h2_h, 256, 1024, 512, 1.0f, 1);
  gemm_bf16_wmma<<<dim3(4, 13), 256, 0, stream>>>(
      h2_h, d3wT, d3b, outDec, nullptr, 256, 784, 1024, 1.0f, 2);
}
